// SchNetConv_9216999817563
// MI455X (gfx1250) — matrix operand order repaired and checked by reading, not compiled, not run on hardware
//
#include <hip/hip_runtime.h>
#include <hip/hip_bf16.h>

// ---------------- WMMA types & helpers (gfx1250, wave32) ----------------
typedef __attribute__((ext_vector_type(16))) _Float16 v16h;
typedef __attribute__((ext_vector_type(8)))  _Float16 v8h;
typedef __attribute__((ext_vector_type(8)))  float    v8f;
typedef __attribute__((ext_vector_type(4)))  float    v4f;

// A-matrix 16x32 f16 fragment from row-major f16 LDS tile (stride in halfs).
// Lane l: M = l&15, hs = l>>4; halfs[0..7] = K in [kb+hs*8,+8),
// halfs[8..15] = K in [kb+16+hs*8,+8).  (ISA 7.12.2, 16-bit A 16x32)
__device__ __forceinline__ v16h ldsA(const _Float16* t, int stride, int rbase,
                                     int kb, int lane) {
  int m = lane & 15, hs = lane >> 4;
  const _Float16* p = t + (rbase + m) * stride + kb + hs * 8;
  v8h lo = *(const v8h*)p;
  v8h hi = *(const v8h*)(p + 16);
  v16h a;
#pragma unroll
  for (int i = 0; i < 8; ++i) { a[i] = lo[i]; a[i + 8] = hi[i]; }
  return a;
}

// Same fragment but sourced from an f32 LDS tile (convert once at build).
__device__ __forceinline__ v16h ldsA_f32(const float* t, int stride, int rbase,
                                         int kb, int lane) {
  int m = lane & 15, hs = lane >> 4;
  const float* p = t + (rbase + m) * stride + kb + hs * 8;
  v4f q0 = *(const v4f*)p;
  v4f q1 = *(const v4f*)(p + 4);
  v4f q2 = *(const v4f*)(p + 16);
  v4f q3 = *(const v4f*)(p + 20);
  v16h a;
#pragma unroll
  for (int i = 0; i < 4; ++i) {
    a[i]      = (_Float16)q0[i];
    a[4 + i]  = (_Float16)q1[i];
    a[8 + i]  = (_Float16)q2[i];
    a[12 + i] = (_Float16)q3[i];
  }
  return a;
}

// B-matrix 32x16 f16 fragment from K-major (transposed) LDS weight image:
// WT[n*stride + k].  Lane l: N = l&15, halfs[0..7] = K in [kb+hs*8,+8), halfs[8..15] = K in [kb+16+hs*8,+8).
__device__ __forceinline__ v16h ldsB(const _Float16* wt, int stride, int nbase,
                                     int kb, int lane) {
  int n = lane & 15, hs = lane >> 4;
  const _Float16* p = wt + (nbase + n) * stride + kb + hs * 8;
  v8h lo = *(const v8h*)p;
  v8h hi = *(const v8h*)(p + 16);
  v16h b;
#pragma unroll
  for (int i = 0; i < 8; ++i) { b[i] = lo[i]; b[i + 8] = hi[i]; }
  return b;
}

__device__ __forceinline__ v8f wmma_f16(v16h a, v16h b, v8f c) {
  return __builtin_amdgcn_wmma_f32_16x16x32_f16(false, a, false, b,
                                                (short)0, c, false, false);
}

// fast swish: x * rcp(1+exp(-x))  (v_exp_f32 + v_rcp_f32, no IEEE divide)
__device__ __forceinline__ float swishf(float x) {
  return x * __builtin_amdgcn_rcpf(1.0f + __expf(-x));
}

#define F 128   // n_filters == n_atom_basis
#define NB 25   // n_basis

// ---------------- zero workspace ----------------
__global__ void zero_f32(float* __restrict__ p, int n) {
  int i = blockIdx.x * blockDim.x + threadIdx.x;
  if (i < n) p[i] = 0.0f;
}

// ---------------- generic [rows,128] @ [128,128] (+bias, +swish) ----------------
template <bool DOSWISH>
__global__ void rowgemm128(const float* __restrict__ in,
                           const float* __restrict__ W,      // [128][128] row-major
                           const float* __restrict__ bias,   // [128] or null
                           float* __restrict__ out, int nRows) {
  __shared__ _Float16 sWT[F * F];   // W^T, K-major per column  (32 KB)
  __shared__ _Float16 sA[64 * F];   // activation tile f16      (16 KB)
  __shared__ float    sB[F];

  const int tid = threadIdx.x;
  for (int idx = tid; idx < F * F; idx += 128) {
    int n = idx >> 7, k = idx & 127;
    sWT[idx] = (_Float16)W[k * F + n];
  }
  if (tid < F) sB[tid] = bias ? bias[tid] : 0.0f;
  __syncthreads();

  const int lane = tid & 31, wave = tid >> 5;
  const int rb = wave * 16, nn = lane & 15, hs = lane >> 4;
  const int nChunks = (nRows + 63) >> 6;

  for (int c = blockIdx.x; c < nChunks; c += gridDim.x) {
    const int rowBase = c << 6;
    for (int idx = tid; idx < 64 * F; idx += 128) {
      int r = idx >> 7, col = idx & 127;
      int g = rowBase + r;
      sA[idx] = (_Float16)((g < nRows) ? in[(size_t)g * F + col] : 0.0f);
    }
    __syncthreads();

    // hoist A-fragments (invariant across output-column tiles)
    v16h a0 = ldsA(sA, F, rb, 0,  lane);
    v16h a1 = ldsA(sA, F, rb, 32, lane);
    v16h a2 = ldsA(sA, F, rb, 64, lane);
    v16h a3 = ldsA(sA, F, rb, 96, lane);

    const int row0  = rb + hs * 8;
    const int remain = nRows - (rowBase + row0);

#pragma unroll
    for (int jt = 0; jt < 8; ++jt) {
      v8f acc = {};
      acc = wmma_f16(a0, ldsB(sWT, F, jt * 16, 0,  lane), acc);
      acc = wmma_f16(a1, ldsB(sWT, F, jt * 16, 32, lane), acc);
      acc = wmma_f16(a2, ldsB(sWT, F, jt * 16, 64, lane), acc);
      acc = wmma_f16(a3, ldsB(sWT, F, jt * 16, 96, lane), acc);
      const int col = jt * 16 + nn;
      const float bb = sB[col];
      float* p = out + (size_t)(rowBase + row0) * F + col;
#pragma unroll
      for (int r = 0; r < 8; ++r) {
        if (r < remain) {
          float v = acc[r] + bb;
          if (DOSWISH) v = swishf(v);
          p[r * F] = v;
        }
      }
    }
    __syncthreads();
  }
}

// ---------------- fused CFConv edge kernel ----------------
// Per 64-edge chunk: Wfilt = swish(fij@W1+b1)@W2+b2 via WMMA (kept on-chip),
// then m = x[src]*Wfilt*C scattered with f32 atomics into agg[dst].
// fij streamed HBM -> LDS with gfx1250 async-to-LDS (ASYNCcnt) loads.
__global__ void edge_cfconv(const float* __restrict__ fij,
                            const float* __restrict__ rij,
                            const int*   __restrict__ src,
                            const int*   __restrict__ dst,
                            const float* __restrict__ W1,  // [25][128]
                            const float* __restrict__ b1,
                            const float* __restrict__ W2,  // [128][128]
                            const float* __restrict__ b2,
                            const float* __restrict__ x,   // [N][128]
                            float* __restrict__ agg,       // [N][128]
                            int E) {
  __shared__ _Float16 sW1[F * 32];          // W1^T padded K 25->32  (8 KB)
  __shared__ _Float16 sW2[F * F];           // W2^T                  (32 KB)
  __shared__ float    sB1[F], sB2[F];
  // fij f32 tile [64][32] (8 KB) unioned with hidden f16 tile [64][128] (16 KB):
  // fij is consumed into registers before h is written.
  __shared__ __align__(16) char sBuf[64 * F * 2];
  float*    sFij32 = (float*)sBuf;
  _Float16* sH     = (_Float16*)sBuf;
  __shared__ int   sS[64], sD[64];
  __shared__ float sC[64];

  const int tid = threadIdx.x;
  for (int idx = tid; idx < F * 32; idx += 128) {
    int n = idx >> 5, k = idx & 31;
    sW1[idx] = (_Float16)((k < NB) ? W1[k * F + n] : 0.0f);
  }
  for (int idx = tid; idx < F * F; idx += 128) {
    int n = idx >> 7, k = idx & 127;
    sW2[idx] = (_Float16)W2[k * F + n];
  }
  if (tid < F) { sB1[tid] = b1[tid]; sB2[tid] = b2[tid]; }
  __syncthreads();

  const int lane = tid & 31, wave = tid >> 5;
  const int rb = wave * 16, nn = lane & 15, hs = lane >> 4;
  const int nChunks = (E + 63) >> 6;

  for (int c = blockIdx.x; c < nChunks; c += gridDim.x) {
    const int eBase = c << 6;
    const int rowsHere = min(64, E - eBase);

    // ---- async HBM -> LDS stream of the fij tile (flat rows are contiguous)
    {
      const float* g = fij + (size_t)eBase * NB;
      const int lim = rowsHere * NB;
      for (int idx = tid; idx < lim; idx += 128) {
        int row = idx / NB;
        int k   = idx - row * NB;
        unsigned ldsOff = (unsigned)(uintptr_t)(void*)&sFij32[row * 32 + k];
        unsigned long long ga = (unsigned long long)(uintptr_t)(g + idx);
        asm volatile("global_load_async_to_lds_b32 %0, %1, off"
                     :: "v"(ldsOff), "v"(ga) : "memory");
      }
      // zero the K-pad columns (25..31) with plain DS stores (no conflict)
      for (int idx = tid; idx < 64 * 7; idx += 128) {
        int row = idx / 7;
        int k   = NB + (idx - row * 7);
        sFij32[row * 32 + k] = 0.0f;
      }
      if (rowsHere < 64) {  // ragged tail: zero unused rows
        for (int idx = tid + rowsHere * 32; idx < 64 * 32; idx += 128)
          sFij32[idx] = 0.0f;
      }
    }
    if (tid < 64) {
      int e = eBase + tid;
      if (e < E) {
        sS[tid] = src[e];
        sD[tid] = dst[e];
        float rr = rij[e];
        float cc = 0.5f * (__cosf(rr * 0.628318530717958648f) + 1.0f);
        sC[tid] = (rr < 5.0f) ? cc : 0.0f;
      } else { sS[tid] = 0; sD[tid] = 0; sC[tid] = 0.0f; }
    }
    asm volatile("s_wait_asynccnt 0x0" ::: "memory");
    __syncthreads();

    // ---- phase A: h = swish(fij @ W1 + b1); A-fragment built once (f32->f16)
    v16h aF = ldsA_f32(sFij32, 32, rb, 0, lane);
    __syncthreads();   // everyone done reading sFij32 before sH overwrites it

#pragma unroll
    for (int jt = 0; jt < 8; ++jt) {
      v8f acc = {};
      acc = wmma_f16(aF, ldsB(sW1, 32, jt * 16, 0, lane), acc);
      const int col = jt * 16 + nn;
      const float bb = sB1[col];
#pragma unroll
      for (int r = 0; r < 8; ++r) {
        int row = rb + hs * 8 + r;
        sH[row * F + col] = (_Float16)swishf(acc[r] + bb);
      }
    }
    __syncthreads();

    // ---- phase B: Wfilt = h @ W2 + b2, then gather/scale/scatter
    v16h h0 = ldsA(sH, F, rb, 0,  lane);
    v16h h1 = ldsA(sH, F, rb, 32, lane);
    v16h h2 = ldsA(sH, F, rb, 64, lane);
    v16h h3 = ldsA(sH, F, rb, 96, lane);

    const int row0   = rb + hs * 8;
    const int remain = E - (eBase + row0);

#pragma unroll
    for (int jt = 0; jt < 8; ++jt) {
      v8f acc = {};
      acc = wmma_f16(h0, ldsB(sW2, F, jt * 16, 0,  lane), acc);
      acc = wmma_f16(h1, ldsB(sW2, F, jt * 16, 32, lane), acc);
      acc = wmma_f16(h2, ldsB(sW2, F, jt * 16, 64, lane), acc);
      acc = wmma_f16(h3, ldsB(sW2, F, jt * 16, 96, lane), acc);
      const int col = jt * 16 + nn;
      const float bb = sB2[col];
#pragma unroll
      for (int r = 0; r < 8; ++r) {
        if (r < remain) {
          int row = row0 + r;
          float wf = acc[r] + bb;
          float m  = x[(size_t)sS[row] * F + col] * wf * sC[row];
          unsafeAtomicAdd(&agg[(size_t)sD[row] * F + col], m);
        }
      }
    }
    __syncthreads();   // sH/sFij32 reused next chunk
  }
}

// ---------------- launch ----------------
extern "C" void kernel_launch(void* const* d_in, const int* in_sizes, int n_in,
                              void* d_out, int out_size, void* d_ws, size_t ws_size,
                              hipStream_t stream) {
  const float* feat   = (const float*)d_in[0];
  const float* fij    = (const float*)d_in[1];
  const float* rij    = (const float*)d_in[2];
  const int*   src    = (const int*)d_in[3];
  const int*   dst    = (const int*)d_in[4];
  const float* W_in2f = (const float*)d_in[5];
  const float* W_f1   = (const float*)d_in[6];
  const float* b_f1   = (const float*)d_in[7];
  const float* W_f2   = (const float*)d_in[8];
  const float* b_f2   = (const float*)d_in[9];
  const float* W_out  = (const float*)d_in[10];
  const float* b_out  = (const float*)d_in[11];

  const int N = in_sizes[0] / F;   // 50000
  const int E = in_sizes[2];       // 1600000

  float* x   = (float*)d_ws;                 // [N][128]  (25.6 MB)
  float* agg = x + (size_t)N * F;            // [N][128]  (25.6 MB)

  // 1) agg = 0  (must re-zero every call: graphs replay without re-poison)
  {
    int n = N * F;
    zero_f32<<<(n + 255) / 256, 256, 0, stream>>>(agg, n);
  }
  // 2) x = feat @ W_in2f (no bias, no activation)
  {
    int blocks = (N + 63) / 64;
    rowgemm128<false><<<blocks, 128, 0, stream>>>(feat, W_in2f, nullptr, x, N);
  }
  // 3) fused filter-MLP + cutoff + gather + scatter-add
  {
    int nChunks = (E + 63) / 64;
    int blocks = nChunks < 2048 ? nChunks : 2048;   // persistent, grid-stride
    edge_cfconv<<<blocks, 128, 0, stream>>>(fij, rij, src, dst, W_f1, b_f1,
                                            W_f2, b_f2, x, agg, E);
  }
  // 4) out = swish(agg @ W_out + b_out)
  {
    int blocks = (N + 63) / 64;
    rowgemm128<true><<<blocks, 128, 0, stream>>>(agg, W_out, b_out,
                                                 (float*)d_out, N);
  }
}
